// FullFastSTU_15753940042017
// MI455X (gfx1250) — matrix operand order repaired and checked by reading, not compiled, not run on hardware
//
#include <hip/hip_runtime.h>
#include <hip/hip_bf16.h>

typedef __attribute__((ext_vector_type(16))) __bf16 v16bf;
typedef __attribute__((ext_vector_type(8)))  __bf16 v8bf;
typedef __attribute__((ext_vector_type(8)))  float  v8f;

#define BSZ   2
#define T_    1024
#define D_    768
#define KEIG  24
#define S_    128
#define KX_   5
#define MROWS (BSZ * T_)   // 2048

// ---------------- Prep 1: bf16 conversions (x row-major, M_inputs transposed) ----------------
__global__ void prep_convert(const float* __restrict__ x, const float* __restrict__ Mi,
                             __bf16* __restrict__ xbf, __bf16* __restrict__ MbfT) {
    int tid = blockIdx.x * blockDim.x + threadIdx.x;
    int stride = gridDim.x * blockDim.x;
    const int totx = MROWS * D_;
    for (int i = tid; i < totx; i += stride)
        xbf[i] = (__bf16)x[i];
    const int totm = D_ * D_;
    for (int i = tid; i < totm; i += stride) {
        int dcol = i / D_, crow = i % D_;          // MbfT[d][c] = M_inputs[c][d]
        MbfT[dcol * D_ + crow] = (__bf16)Mi[crow * D_ + dcol];
    }
}

// ---------------- Prep 2: fused per-channel readout weights ----------------
__global__ void prep_w(const float* __restrict__ Mf, const float* __restrict__ C,
                       const float* __restrict__ Mdir,
                       float* __restrict__ w, float* __restrict__ wdir) {
    int d = blockIdx.x;      // 0..767
    int s = threadIdx.x;     // 0..127
    float acc = 0.f;
    for (int k = 0; k < KEIG; ++k) {
        float mf = Mf[k * D_ + d];
        acc = fmaf(mf, C[k * S_ + s] + C[(KEIG + k) * S_ + s], acc);
    }
    w[d * S_ + s] = acc;
    if (s < KX_) {
        float a = 0.f;
        for (int k = 0; k < KEIG; ++k)
            a = fmaf(Mf[k * D_ + d], Mdir[k * KX_ + s] + Mdir[(KEIG + k) * KX_ + s], a);
        wdir[d * 8 + s] = a;
    }
}

// ---------------- GEMM: u[(b*D+d), t] = (x @ M_inputs)[b,t,d]  via bf16 WMMA ----------------
// Block = 256 threads = 8 waves; each wave owns one 16x16 output tile; block tile = 32(M) x 64(N).
__global__ __launch_bounds__(256) void gemm_u(const __bf16* __restrict__ xbf,
                                              const __bf16* __restrict__ MbfT,
                                              float* __restrict__ u) {
    int wave = threadIdx.x >> 5;
    int lane = threadIdx.x & 31;
    int mTile = blockIdx.x * 2 + (wave & 1);
    int nTile = blockIdx.y * 4 + (wave >> 1);
    int row0 = mTile * 16;
    int col0 = nTile * 16;
    int l15 = lane & 15, hi = lane >> 4;

    // A fragment (16x32 bf16): lane (l15,hi) holds row m=l15, K = hi*16 .. hi*16+15
    const __bf16* arow = xbf + (size_t)(row0 + l15) * D_;
    // B fragment (32x16 bf16): lane (l15,hi) holds col n=l15, K = hi*16 .. hi*16+15 (consecutive in MbfT)
    const __bf16* brow = MbfT + (size_t)(col0 + l15) * D_ + hi * 16;

    v8f acc = {};
    for (int kk = 0; kk < D_; kk += 32) {
        v8bf a0 = *(const v8bf*)(arow + kk + hi * 16);
        v8bf a1 = *(const v8bf*)(arow + kk + hi * 16 + 8);
        v16bf a;
#pragma unroll
        for (int i = 0; i < 8; ++i) { a[i] = a0[i]; a[8 + i] = a1[i]; }
        v16bf b = *(const v16bf*)(brow + kk);
        acc = __builtin_amdgcn_wmma_f32_16x16x32_bf16(
            /*neg_a=*/false, a, /*neg_b=*/false, b,
            /*c_mod=*/(short)0, acc, /*reuse_a=*/false, /*reuse_b=*/false);
    }

    // D layout: acc[r] = D[M = r + 8*hi][N = l15]; write transposed into u[(b*D+d)*T + t]
    int d = col0 + l15;
    int grow = row0 + 8 * hi;          // global M row of acc[0]
    int b = grow / T_;
    int t = grow % T_;                 // 16-row tiles never straddle batch boundary (T_%16==0)
    float* dst = u + ((size_t)(b * D_ + d)) * T_ + t;
#pragma unroll
    for (int r = 0; r < 8; ++r) dst[r] = acc[r];
}

// ---------------- Scan: diagonal LDS recurrence + fused readout + direct taps ----------------
// One block per sequence n (=b*D+d), 128 threads = 128 states, 4 waves.
__global__ __launch_bounds__(128) void scan_out(const float* __restrict__ u,
                                                const float* __restrict__ A,
                                                const float* __restrict__ Bm,
                                                const float* __restrict__ w,
                                                const float* __restrict__ wdir,
                                                __hip_bfloat16* __restrict__ out) {
    __shared__ float uch[128];
    __shared__ float acc[128];
    int n = blockIdx.x;
    int d = n % D_;
    int b = n / D_;
    int s = threadIdx.x;
    int lane = s & 31;

    float a_s  = A[s];
    float bm_s = Bm[s];
    float w_s  = w[d * S_ + s];
    float wd[KX_];
#pragma unroll
    for (int j = 0; j < KX_; ++j) wd[j] = wdir[d * 8 + j];

    const float* useq = u + (size_t)n * T_;
    float h = 0.f;

    for (int c = 0; c < T_; c += 128) {
        uch[s] = useq[c + s];
        acc[s] = 0.f;
        __syncthreads();
        for (int i = 0; i < 128; ++i) {
            float ut = uch[i];
            h = fmaf(a_s, h, bm_s * ut);     // h_t = A*h_{t-1} + Bm*u_t
            float p = w_s * h;               // fused readout partial
            p += __shfl_xor(p, 16);
            p += __shfl_xor(p, 8);
            p += __shfl_xor(p, 4);
            p += __shfl_xor(p, 2);
            p += __shfl_xor(p, 1);
            if (lane == 0) atomicAdd(&acc[i], p);   // ds_add_f32, 4 adds per timestep
        }
        __syncthreads();
        // finalize timestep t = c + s: add causal 5-tap direct term, emit bf16
        int t = c + s;
        float val = acc[s];
#pragma unroll
        for (int j = 0; j < KX_; ++j) {
            int tj = t - j;
            float uu = (s - j >= 0) ? uch[s - j] : ((tj >= 0) ? useq[tj] : 0.f);
            val = fmaf(wd[j], uu, val);
        }
        out[((size_t)(b * T_ + t)) * D_ + d] = __float2bfloat16(val);
        __syncthreads();
    }
}

// ---------------- Launch ----------------
extern "C" void kernel_launch(void* const* d_in, const int* in_sizes, int n_in,
                              void* d_out, int out_size, void* d_ws, size_t ws_size,
                              hipStream_t stream) {
    const float* x    = (const float*)d_in[0];
    // d_in[1] = input_pos (unused)
    const float* Mi   = (const float*)d_in[2];
    const float* Mf   = (const float*)d_in[3];
    const float* A    = (const float*)d_in[4];
    const float* Bm   = (const float*)d_in[5];
    const float* C    = (const float*)d_in[6];
    const float* Mdir = (const float*)d_in[7];
    __hip_bfloat16* out = (__hip_bfloat16*)d_out;

    char* ws = (char*)d_ws;
    float*  u    = (float*)(ws + 0);               // 1536*1024 f32   = 6,291,456 B
    __bf16* xbf  = (__bf16*)(ws + 6291456);        // 2048*768 bf16   = 3,145,728 B
    __bf16* MbfT = (__bf16*)(ws + 9437184);        // 768*768 bf16    = 1,179,648 B
    float*  w    = (float*)(ws + 10616832);        // 768*128 f32     =   393,216 B
    float*  wdir = (float*)(ws + 11010048);        // 768*8 f32       =    24,576 B

    prep_convert<<<512, 256, 0, stream>>>(x, Mi, xbf, MbfT);
    prep_w<<<D_, S_, 0, stream>>>(Mf, C, Mdir, w, wdir);
    gemm_u<<<dim3(MROWS / 32, D_ / 64), 256, 0, stream>>>(xbf, MbfT, u);
    scan_out<<<BSZ * D_, S_, 0, stream>>>(u, A, Bm, w, wdir, out);
}
